// RelPosRFFBias_412316860881
// MI455X (gfx1250) — matrix operand order repaired and checked by reading, not compiled, not run on hardware
//
#include <hip/hip_runtime.h>
#include <hip/hip_bf16.h>
#include <math.h>

typedef __attribute__((ext_vector_type(16))) _Float16 v16h;
typedef __attribute__((ext_vector_type(8)))  _Float16 v8h;
typedef __attribute__((ext_vector_type(8)))  float    v8f;

#define TT      512
#define NB      8
#define NHEAD   16
#define HID     64
#define INV2PI  0.15915494309189535f

// Exact-erf gelu: 0.5*x*(1+erf(x/sqrt2)) via Abramowitz-Stegun 7.1.26
// (|erf err| <= 1.5e-7, far below the f16 WMMA quantization already in play).
// Branchless: ~10 FMA/MUL + v_rcp + v_exp. The 0.5 of gelu is folded into the
// polynomial coefficients. Raw v_exp_f32 is safe: the argument is
// -z^2*log2(e) which for any realistic pre-activation stays far above the
// -126 denormal threshold, and underflow-to-zero is the correct limit anyway.
__device__ __forceinline__ float gelu_exact(float x) {
    const float z  = fabsf(x) * 0.70710678118654752f;            // |x|/sqrt2
    const float t  = __builtin_amdgcn_rcpf(fmaf(0.3275911f, z, 1.0f));
    float p = fmaf(0.5f *  1.061405429f, t, 0.5f * -1.453152027f);
    p = fmaf(p, t, 0.5f *  1.421413741f);
    p = fmaf(p, t, 0.5f * -0.284496736f);
    p = fmaf(p, t, 0.5f *  0.254829592f);
    p = p * t;
    const float e  = __builtin_amdgcn_exp2f(z * z * -1.4426950408889634f); // e^{-z^2}
    const float hw = p * e;                                      // 0.5*(1-erf(z))
    const float gh = (x >= 0.0f) ? (1.0f - hw) : hw;             // 0.5*(1+erf(x/sqrt2))
    return x * gh;
}

__global__ __launch_bounds__(256)
void rff_bias_kernel(const float* __restrict__ centers,
                     const unsigned char* __restrict__ mask,
                     const float* __restrict__ phase,
                     const float* __restrict__ W1,
                     const float* __restrict__ b1,
                     const float* __restrict__ W2,
                     const float* __restrict__ b2,
                     float* __restrict__ out)
{
    const int b  = blockIdx.x;      // batch
    const int tt = blockIdx.y;      // t-tile index (16 rows)
    const int ss = blockIdx.z;      // s-tile index (16 cols)
    if (ss < tt) return;            // symmetry: only upper-triangular macro-tiles
    const int T0 = tt << 4;
    const int S0 = ss << 4;

    __shared__ float s_cT[16], s_cS[16], s_mT[16], s_mS[16];
    __shared__ __align__(16) _Float16 s_h[8][16 * HID];      // per-wave h[pair][hidden]
    __shared__ __align__(16) float s_tr[16][NHEAD][16];      // [s_local][head][t_local]

    if (threadIdx.x < 16) {
        s_cT[threadIdx.x] = centers[b * TT + T0 + threadIdx.x];
        s_mT[threadIdx.x] = mask[b * TT + T0 + threadIdx.x] ? 1.0f : 0.0f;
    } else if (threadIdx.x < 32) {
        const int i = threadIdx.x - 16;
        s_cS[i] = centers[b * TT + S0 + i];
        s_mS[i] = mask[b * TT + S0 + i] ? 1.0f : 0.0f;
    }
    __syncthreads();

    const int lane = threadIdx.x & 31;
    const int wave = threadIdx.x >> 5;
    const int l    = lane & 15;   // N-index (head / hidden col / pair col)
    const int hi   = lane >> 4;   // which K-half this lane owns

    // ---- Constant WMMA operands, register-resident per lane ----
    // GEMM-1 B tiles: W1 [32 feats][64 hidden] row-major.
    v16h w1b[4];
    float b1v[4];
#pragma unroll
    for (int n = 0; n < 4; ++n) {
#pragma unroll
        for (int j = 0; j < 16; ++j)
            w1b[n][j] = (_Float16)W1[((j & 8) * 2 + 8 * hi + (j & 7)) * HID + 16 * n + l];
        b1v[n] = b1[16 * n + l];
    }
    // GEMM-2 B tiles: W2 [64 hidden][16 heads] row-major, K=64 split as 2x32.
    v16h w2b[2];
#pragma unroll
    for (int w = 0; w < 2; ++w)
#pragma unroll
        for (int j = 0; j < 16; ++j)
            w2b[w][j] = (_Float16)W2[(32 * w + (j & 8) * 2 + 8 * hi + (j & 7)) * NHEAD + l];
    const float b2v = b2[l];

    // Per-lane RFF constants: freqs = logspace(log10 2, log10 64, 16) = 2^(1+k/3).
    float fr[8], ph[8];
#pragma unroll
    for (int j = 0; j < 8; ++j) {
        const int k = 8 * hi + j;
        fr[j] = __builtin_amdgcn_exp2f(1.0f + (float)k * (1.0f / 3.0f));
        ph[j] = phase[k] * INV2PI;     // v_sin/v_cos take revolutions
    }

    _Float16* const hb = &s_h[wave][0];

    // Each wave computes 2 t-rows of the 16x16 macro-tile (8 waves x 2 = 16).
    for (int g = 0; g < 2; ++g) {
        const int   tl = (wave << 1) + g;
        const float ct = s_cT[tl];
        const float mt = s_mT[tl];
        const float d  = fabsf(ct - s_cS[l]);            // pair M = s_local = l

        // f16 A-matrix (16 pairs x 32 feats), built entirely lane-locally:
        // lane (l,hi): a[j] = sin feat K=8*hi+j ; a[j+8] = cos feat K=16+8*hi+j.
        v16h a;
#pragma unroll
        for (int j = 0; j < 8; ++j) {
            const float rev = d * fr[j] + ph[j];
            a[j]     = (_Float16)__builtin_amdgcn_sinf(rev);
            a[j + 8] = (_Float16)__builtin_amdgcn_cosf(rev);
        }

        // ---- GEMM-1: h = gelu(feats @ W1 + b1), 4 N-tiles of 16 hidden ----
#pragma unroll
        for (int n = 0; n < 4; ++n) {
            v8f c;
#pragma unroll
            for (int r = 0; r < 8; ++r) c[r] = b1v[n];
            v8f hacc = __builtin_amdgcn_wmma_f32_16x16x32_f16(
                false, a, false, w1b[n], (short)0, c, false, false);
#pragma unroll
            for (int r = 0; r < 8; ++r)
                hb[(r + 8 * hi) * HID + 16 * n + l] = (_Float16)gelu_exact(hacc[r]);
        }
        __builtin_amdgcn_wave_barrier();   // order LDS stores before cross-lane reads

        // ---- GEMM-2: bias = h @ W2 + b2, two K=32 WMMAs ----
        v8f acc;
#pragma unroll
        for (int r = 0; r < 8; ++r) acc[r] = b2v;
#pragma unroll
        for (int w = 0; w < 2; ++w) {
            const v8h* row = reinterpret_cast<const v8h*>(hb + l * HID + 32 * w + 8 * hi);
            const v8h lo = row[0];
            const v8h hh = row[2];
            v16h a2;
#pragma unroll
            for (int j = 0; j < 8; ++j) { a2[j] = lo[j]; a2[j + 8] = hh[j]; }
            acc = __builtin_amdgcn_wmma_f32_16x16x32_f16(
                false, a2, false, w2b[w], (short)0, acc, false, false);
        }
        __builtin_amdgcn_wave_barrier();   // next g's LDS writes stay after these reads

        // ---- Masked values; normal (coalesced) store + transpose staging ----
        const int sb = 8 * hi;
        float v[8];
#pragma unroll
        for (int r = 0; r < 8; ++r) v[r] = acc[r] * mt * s_mS[sb + r];

        float* const orow = out + (((size_t)(b * NHEAD + l) * TT + (T0 + tl)) * TT) + S0 + sb;
        reinterpret_cast<float4*>(orow)[0] = make_float4(v[0], v[1], v[2], v[3]);
        reinterpret_cast<float4*>(orow)[1] = make_float4(v[4], v[5], v[6], v[7]);

        // stage for the mirrored tile: s_tr[s_local][head][t_local]
#pragma unroll
        for (int r = 0; r < 8; ++r) s_tr[sb + r][l][tl] = v[r];
    }

    __syncthreads();   // all waves' staging visible block-wide

    // ---- Mirror store: bias[b,h,S0+sl,T0..T0+15], rows contiguous = b128 ----
    for (int g = 0; g < 2; ++g) {
        const int sl = (wave << 1) + g;
        const float* src = &s_tr[sl][l][8 * hi];
        const float4 m0 = reinterpret_cast<const float4*>(src)[0];
        const float4 m1 = reinterpret_cast<const float4*>(src)[1];
        float* const mrow = out + (((size_t)(b * NHEAD + l) * TT + (S0 + sl)) * TT) + T0 + 8 * hi;
        reinterpret_cast<float4*>(mrow)[0] = m0;
        reinterpret_cast<float4*>(mrow)[1] = m1;
    }
}

extern "C" void kernel_launch(void* const* d_in, const int* in_sizes, int n_in,
                              void* d_out, int out_size, void* d_ws, size_t ws_size,
                              hipStream_t stream) {
    const float*         centers = (const float*)d_in[0];
    const unsigned char* mask    = (const unsigned char*)d_in[1];
    const float*         phase   = (const float*)d_in[2];
    const float*         W1      = (const float*)d_in[3];
    const float*         b1      = (const float*)d_in[4];
    const float*         W2      = (const float*)d_in[5];
    const float*         b2      = (const float*)d_in[6];
    float* out = (float*)d_out;

    // One block per (batch, 16x16 macro-tile); only ss>=tt blocks do work,
    // each writes its tile and the mirrored tile (both as coalesced rows).
    dim3 grid(NB, TT / 16, TT / 16);
    dim3 block(256);
    hipLaunchKernelGGL(rff_bias_kernel, grid, block, 0, stream,
                       centers, mask, phase, W1, b1, W2, b2, out);
}
